// GATInteraction_996432412735
// MI455X (gfx1250) — matrix operand order repaired and checked by reading, not compiled, not run on hardware
//
#include <hip/hip_runtime.h>
#include <hip/hip_bf16.h>

#define H 3
#define D 128
#define N_ADS 16384
#define N_CAT 65536
#define NN (N_ADS + N_CAT)        // 81920 nodes
#define EE 524288
#define HD (H * D)                 // 384
#define NEG_SLOPE 0.2f
#define EPS_SM 1e-16f
#define EPS_NORM 1e-12f
#define M_SUB 4                    // M-tiles per block in the WMMA GEMM

typedef __attribute__((ext_vector_type(16))) __bf16   v16bf;
typedef __attribute__((ext_vector_type(8)))  float    v8f;
typedef __attribute__((ext_vector_type(4)))  float    v4f;
typedef __attribute__((ext_vector_type(4)))  unsigned int v4u;

// ---------- helpers ----------
__device__ __forceinline__ unsigned short f32_to_bf16(float f) {
    unsigned int u = __float_as_uint(f);
    unsigned int r = (u + 0x7FFFu + ((u >> 16) & 1u)) >> 16;   // round-nearest-even
    return (unsigned short)r;
}

__device__ __forceinline__ void atomicMaxF(float* addr, float v) {
    // sign-split trick: works with init = -inf
    if (v >= 0.0f) atomicMax((int*)addr, __float_as_int(v));
    else           atomicMin((unsigned int*)addr, (unsigned int)__float_as_int(v));
}

// ---------- 1) convert x (combined) to bf16 and W to bf16-TRANSPOSED ----------
// wbfT is [HD x D] row-major:  wbfT[n*D + k] = bf16(W[k*HD + n])
__global__ void k_convert(const float* __restrict__ h_ads,
                          const float* __restrict__ h_cat,
                          const float* __restrict__ W,
                          unsigned short* __restrict__ xbf,
                          unsigned short* __restrict__ wbfT) {
    long long i = (long long)blockIdx.x * blockDim.x + threadIdx.x;
    const long long nx = (long long)NN * D;
    if (i < nx) {
        float v = (i < (long long)N_ADS * D) ? h_ads[i] : h_cat[i - (long long)N_ADS * D];
        xbf[i] = f32_to_bf16(v);
    } else {
        long long j = i - nx;                 // j indexes W row-major [D x HD]
        if (j < (long long)D * HD) {
            const int k = (int)(j / HD);
            const int n = (int)(j % HD);
            wbfT[(long long)n * D + k] = f32_to_bf16(W[j]);
        }
    }
}

// ---------- 2) init: softmax-max = -inf, denom = 0, accumulator = 0 ----------
__global__ void k_init(float* __restrict__ m, float* __restrict__ denom,
                       float* __restrict__ acc) {
    long long i = (long long)blockIdx.x * blockDim.x + threadIdx.x;
    if (i < (long long)NN * D) acc[i] = 0.0f;
    if (i < (long long)NN * H) { m[i] = -__builtin_inff(); denom[i] = 0.0f; }
}

// ---------- 3) WMMA GEMM: xp[N,384] = x[N,128] @ W[128,384] (bf16 in, f32 acc) ----
// Block = 8 waves; wave covers 3 N-tiles (all 24 across block) x M_SUB M-tiles.
// B tiles loaded once per K-step and reused across the 4 M-tiles (register reuse).
__global__ void __launch_bounds__(256)
k_gemm_wmma(const unsigned short* __restrict__ xbf,
            const unsigned short* __restrict__ wbfT,
            float* __restrict__ xp) {
    const int lane    = threadIdx.x & 31;
    const int wave    = threadIdx.x >> 5;
    const int lid     = lane & 15;
    const int halfSel = lane >> 4;                 // 0: lanes 0-15, 1: lanes 16-31
    const int mBase   = blockIdx.x * (16 * M_SUB); // 1280 blocks * 64 rows = 81920
    const int nBase   = wave * 48;                 // 8 waves * 3 tiles * 16 = 384

    union VU { v4u u[2]; v16bf b; };

    v8f acc[M_SUB][3];
    #pragma unroll
    for (int m = 0; m < M_SUB; ++m)
        #pragma unroll
        for (int t = 0; t < 3; ++t)
            acc[m][t] = (v8f){0.f,0.f,0.f,0.f,0.f,0.f,0.f,0.f};

    #pragma unroll
    for (int k0 = 0; k0 < D; k0 += 32) {
        // B tiles 32x16 bf16: lane-half -> N=lid, K=k0+8h..+7 & k0+16+8h..+7 (matches A).
        // W stored transposed -> 2x 16-byte b128 loads per tile.
        const int kB = k0 + halfSel * 8;
        v16bf B[3];
        #pragma unroll
        for (int t = 0; t < 3; ++t) {
            VU bu;
            const unsigned short* brow = wbfT + (long long)(nBase + t * 16 + lid) * D + kB;
            bu.u[0] = *(const v4u*)(brow);
            bu.u[1] = *(const v4u*)(brow + 16);
            B[t] = bu.b;
        }
        // A tiles 16x32 bf16 (ISA 7.12.2): lane<16 -> M=lid, K=k0..+7 & k0+16..+23
        //                                  lane>=16 -> M=lid, K=k0+8..15 & k0+24..31
        const int kA = k0 + halfSel * 8;
        #pragma unroll
        for (int m = 0; m < M_SUB; ++m) {
            VU au;
            const unsigned short* arow = xbf + (long long)(mBase + m * 16 + lid) * D + kA;
            au.u[0] = *(const v4u*)(arow);
            au.u[1] = *(const v4u*)(arow + 16);
            v16bf A = au.b;
            #pragma unroll
            for (int t = 0; t < 3; ++t)
                acc[m][t] = __builtin_amdgcn_wmma_f32_16x16x32_bf16(
                        false, A, false, B[t], (short)0, acc[m][t], false, false);
        }
    }

    // C/D layout: VGPR i -> row (halfSel*8 + i), col = n0 + lid
    #pragma unroll
    for (int m = 0; m < M_SUB; ++m) {
        #pragma unroll
        for (int t = 0; t < 3; ++t) {
            const int col = nBase + t * 16 + lid;
            #pragma unroll
            for (int i = 0; i < 8; ++i) {
                const int row = mBase + m * 16 + halfSel * 8 + i;
                xp[(long long)row * HD + col] = acc[m][t][i];
            }
        }
    }
}

// ---------- 4) attention logits: al[n,h] = <xp[n,h,:], a[h,:]> ----------
__global__ void k_attn_logits(const float* __restrict__ xp,
                              const float* __restrict__ a_src,
                              const float* __restrict__ a_dst,
                              float* __restrict__ alsrc,
                              float* __restrict__ aldst) {
    const int lane = threadIdx.x & 31;
    const int n = blockIdx.x * (blockDim.x >> 5) + (threadIdx.x >> 5);
    if (n >= NN) return;
    #pragma unroll
    for (int h = 0; h < H; ++h) {
        const v4f v  = *(const v4f*)(xp + (long long)n * HD + h * D + lane * 4);
        const v4f as = *(const v4f*)(a_src + h * D + lane * 4);
        const v4f ad = *(const v4f*)(a_dst + h * D + lane * 4);
        float ps = v[0]*as[0] + v[1]*as[1] + v[2]*as[2] + v[3]*as[3];
        float pd = v[0]*ad[0] + v[1]*ad[1] + v[2]*ad[2] + v[3]*ad[3];
        #pragma unroll
        for (int off = 16; off; off >>= 1) {
            ps += __shfl_xor(ps, off, 32);
            pd += __shfl_xor(pd, off, 32);
        }
        if (lane == 0) { alsrc[n * H + h] = ps; aldst[n * H + h] = pd; }
    }
}

// ---------- 5) edge pass 1: leaky-relu score + segment max ----------
__global__ void k_edge_max(const long long* __restrict__ ei,
                           const float* __restrict__ alsrc,
                           const float* __restrict__ aldst,
                           float* __restrict__ score,
                           float* __restrict__ m) {
    const long long e = (long long)blockIdx.x * blockDim.x + threadIdx.x;
    if (e >= EE) return;
    const long long s = ei[e], d = ei[EE + e];
    #pragma unroll
    for (int h = 0; h < H; ++h) {
        float sc = alsrc[s * H + h] + aldst[d * H + h];
        sc = (sc > 0.f) ? sc : NEG_SLOPE * sc;
        score[e * H + h] = sc;
        atomicMaxF(m + d * H + h, sc);
    }
}

// ---------- 6) edge pass 2: exp(e - max) + segment sum ----------
__global__ void k_edge_exp(const long long* __restrict__ ei,
                           const float* __restrict__ m,
                           float* __restrict__ score,
                           float* __restrict__ denom) {
    const long long e = (long long)blockIdx.x * blockDim.x + threadIdx.x;
    if (e >= EE) return;
    const long long d = ei[EE + e];
    #pragma unroll
    for (int h = 0; h < H; ++h) {
        const float ex = __expf(score[e * H + h] - m[d * H + h]);
        score[e * H + h] = ex;
        atomicAdd(denom + d * H + h, ex);
    }
}

// ---------- 7) edge pass 3: out[dst] += mean_h(alpha * xp[src,h,:]) ----------
__global__ void k_edge_agg(const long long* __restrict__ ei,
                           const float* __restrict__ score,
                           const float* __restrict__ denom,
                           const float* __restrict__ xp,
                           float* __restrict__ acc) {
    const int lane = threadIdx.x & 31;
    const long long e = (long long)blockIdx.x * (blockDim.x >> 5) + (threadIdx.x >> 5);
    if (e >= EE) return;
    const long long s = ei[e], d = ei[EE + e];
    const float a0 = score[e * H + 0] / (denom[d * H + 0] + EPS_SM);
    const float a1 = score[e * H + 1] / (denom[d * H + 1] + EPS_SM);
    const float a2 = score[e * H + 2] / (denom[d * H + 2] + EPS_SM);
    const float inv3 = (1.0f / 3.0f);
    const float* xr = xp + (long long)s * HD;
    // lane-contiguous float4: 3x b128 loads per lane, wave stays fully coalesced
    const v4f x0 = *(const v4f*)(xr + lane * 4);
    const v4f x1 = *(const v4f*)(xr + D + lane * 4);
    const v4f x2 = *(const v4f*)(xr + 2 * D + lane * 4);
    float* ap = acc + d * D + lane * 4;
    #pragma unroll
    for (int j = 0; j < 4; ++j) {
        const float v = a0 * x0[j] + a1 * x1[j] + a2 * x2[j];
        atomicAdd(ap + j, v * inv3);
    }
}

// ---------- 8) bias + L2 normalize + residual ----------
__global__ void k_finalize(const float* __restrict__ acc,
                           const float* __restrict__ bias,
                           const float* __restrict__ h_ads,
                           const float* __restrict__ h_cat,
                           float* __restrict__ out) {
    const int lane = threadIdx.x & 31;
    const int n = blockIdx.x * (blockDim.x >> 5) + (threadIdx.x >> 5);
    if (n >= NN) return;
    const v4f a = *(const v4f*)(acc + (long long)n * D + lane * 4);
    const v4f b = *(const v4f*)(bias + lane * 4);
    v4f v = a + b;
    float ss = v[0]*v[0] + v[1]*v[1] + v[2]*v[2] + v[3]*v[3];
    #pragma unroll
    for (int off = 16; off; off >>= 1) ss += __shfl_xor(ss, off, 32);
    const float scale = 1.0f / fmaxf(sqrtf(ss), EPS_NORM);
    const v4f orig = (n < N_ADS)
        ? *(const v4f*)(h_ads + (long long)n * D + lane * 4)
        : *(const v4f*)(h_cat + (long long)(n - N_ADS) * D + lane * 4);
    v4f r;
    #pragma unroll
    for (int j = 0; j < 4; ++j) r[j] = v[j] * scale + orig[j];
    *(v4f*)(out + (long long)n * D + lane * 4) = r;
}

extern "C" void kernel_launch(void* const* d_in, const int* in_sizes, int n_in,
                              void* d_out, int out_size, void* d_ws, size_t ws_size,
                              hipStream_t stream) {
    const float*     h_ads = (const float*)d_in[0];
    const float*     h_cat = (const float*)d_in[1];
    const long long* edges = (const long long*)d_in[2];
    const float*     W     = (const float*)d_in[3];
    const float*     a_src = (const float*)d_in[4];
    const float*     a_dst = (const float*)d_in[5];
    const float*     bias  = (const float*)d_in[6];
    float* out = (float*)d_out;

    // workspace carve (256B aligned)
    char* ws = (char*)d_ws;
    size_t off = 0;
    auto carve = [&](size_t bytes) -> char* {
        char* p = ws + off;
        off = (off + bytes + 255) & ~(size_t)255;
        return p;
    };
    float*          xp    = (float*)         carve((size_t)NN * HD * 4);
    unsigned short* xbf   = (unsigned short*)carve((size_t)NN * D * 2);
    unsigned short* wbfT  = (unsigned short*)carve((size_t)D * HD * 2);
    float*          alsrc = (float*)         carve((size_t)NN * H * 4);
    float*          aldst = (float*)         carve((size_t)NN * H * 4);
    float*          mmax  = (float*)         carve((size_t)NN * H * 4);
    float*          denom = (float*)         carve((size_t)NN * H * 4);
    float*          score = (float*)         carve((size_t)EE * H * 4);
    float*          acc   = (float*)         carve((size_t)NN * D * 4);
    (void)ws_size; (void)in_sizes; (void)n_in; (void)out_size;

    const int T = 256;
    // 1) bf16 conversion of x and W^T
    {
        long long total = (long long)NN * D + (long long)D * HD;
        int blocks = (int)((total + T - 1) / T);
        k_convert<<<blocks, T, 0, stream>>>(h_ads, h_cat, W, xbf, wbfT);
    }
    // 2) init max/denom/acc
    {
        long long total = (long long)NN * D;
        int blocks = (int)((total + T - 1) / T);
        k_init<<<blocks, T, 0, stream>>>(mmax, denom, acc);
    }
    // 3) WMMA GEMM: 64 rows per block, B tiles reused across 4 M-tiles
    k_gemm_wmma<<<NN / (16 * M_SUB), T, 0, stream>>>(xbf, wbfT, xp);
    // 4) attention logits (wave per node)
    k_attn_logits<<<NN / (T / 32), T, 0, stream>>>(xp, a_src, a_dst, alsrc, aldst);
    // 5) segment max
    k_edge_max<<<EE / T, T, 0, stream>>>(edges, alsrc, aldst, score, mmax);
    // 6) exp + segment sum
    k_edge_exp<<<EE / T, T, 0, stream>>>(edges, mmax, score, denom);
    // 7) weighted aggregation (wave per edge)
    k_edge_agg<<<EE / (T / 32), T, 0, stream>>>(edges, score, denom, xp, acc);
    // 8) bias + normalize + residual (wave per node)
    k_finalize<<<NN / (T / 32), T, 0, stream>>>(acc, bias, h_ads, h_cat, out);
}
